// Mamba_CTC_encoder_71631464563205
// MI455X (gfx1250) — compile-verified, matrix operand order checked by reading, not run on hardware
//
#include <hip/hip_runtime.h>
#include <hip/hip_bf16.h>

typedef _Float16 f16;
typedef __attribute__((ext_vector_type(16))) _Float16 v16h;
typedef __attribute__((ext_vector_type(8)))  _Float16 v8h;
typedef __attribute__((ext_vector_type(8)))  float    v8f;

#define B_   4
#define T_   5120
#define NH_  512
#define DI_  1024
#define DS_  64
#define DR_  32
#define DC_  4
#define L_   1024
#define NCOL (B_ * L_)   // 4096

__device__ __forceinline__ float siluf(float x)     { return x / (1.f + __expf(-x)); }
__device__ __forceinline__ float softplusf(float x) { return x > 20.f ? x : log1pf(__expf(x)); }

union V16U { v16h v; v8h h[2]; };

// ---------------------------------------------------------------------------
// Generic tiled WMMA GEMM:  C[M,N] = act( (A[M,K] * B[K,N]) * alpha[m] + beta[m] )
// A: f16 row-major weights. B: f16, either dense [K][N] (mode 0) or implicit
// im2col of a feature-major activation tensor (mode 1). C: f16 [M][N].
// Block: 256 threads = 8 waves. Tile: 128(M) x 64(N), K-step 32.
// B tile stored TRANSPOSED in LDS so each lane's WMMA fragment is two aligned
// 16-byte chunks (ds_load_b128). Dense staging uses 16-byte global loads.
// ---------------------------------------------------------------------------
__global__ __launch_bounds__(256) void gemm_wmma_f16(
    const f16* __restrict__ A, const f16* __restrict__ Bsrc, f16* __restrict__ C,
    int M, int N, int K,
    const float* __restrict__ alpha, const float* __restrict__ beta, int act,
    int mode, int ksz, int stride, int pad, int Lin, int Lout)
{
    __shared__ alignas(16) f16 As [128][40];   // [m][k], row stride 80B (16B mult.)
    __shared__ alignas(16) f16 BsT[ 64][40];   // [n][k], transposed B tile

    const int tid  = threadIdx.x;
    const int wave = tid >> 5;
    const int lane = tid & 31;
    const int half = lane >> 4;
    const int l16  = lane & 15;
    const int bm   = blockIdx.y * 128;
    const int bn   = blockIdx.x * 64;
    const int nb   = (mode == 1) ? (N / Lout) : 1;   // batches packed in N

    v8f acc[4];
    #pragma unroll
    for (int i = 0; i < 4; ++i)
        #pragma unroll
        for (int j = 0; j < 8; ++j) acc[i][j] = 0.f;

    for (int kt = 0; kt < K; kt += 32) {
        // --- stage A tile (128 x 32): 2 aligned 16B chunks per thread ---
        #pragma unroll
        for (int c = 0; c < 2; ++c) {
            int chunk = tid + c * 256;          // 0..511
            int r  = chunk >> 2;                // 128 rows, 4 chunks/row
            int k8 = (chunk & 3) * 8;
            int gm = bm + r, gk0 = kt + k8;
            if (gm < M && gk0 + 8 <= K) {
                *(v8h*)&As[r][k8] = *(const v8h*)&A[(size_t)gm * K + gk0];
            } else {
                #pragma unroll
                for (int e = 0; e < 8; ++e)
                    As[r][k8 + e] = (gm < M && gk0 + e < K)
                                        ? A[(size_t)gm * K + gk0 + e] : (f16)0.f;
            }
        }
        // --- stage B tile (32 x 64) -> BsT[n][k] ---
        if (mode == 0) {
            // one aligned 16B global load per thread, transposed LDS store
            int r  = tid >> 3;                  // 32 k-rows, 8 chunks/row
            int c8 = (tid & 7) * 8;
            int gk = kt + r;
            v8h v;
            if (gk < K) {
                v = *(const v8h*)&Bsrc[(size_t)gk * N + bn + c8];
            } else {
                #pragma unroll
                for (int e = 0; e < 8; ++e) v[e] = (f16)0.f;
            }
            #pragma unroll
            for (int e = 0; e < 8; ++e) BsT[c8 + e][r] = v[e];
        } else {
            // implicit im2col: 8 gathered elements per thread, unrolled
            #pragma unroll
            for (int it = 0; it < 8; ++it) {
                int idx = tid + it * 256;       // 0..2047
                int kk = idx >> 6, cc = idx & 63;
                int gk = kt + kk, gn = bn + cc;
                f16 v = (f16)0.f;
                if (gk < K) {
                    int ci = gk / ksz, j = gk - ci * ksz;
                    int bb = gn / Lout, t = gn - bb * Lout;
                    int p  = t * stride - pad + j;
                    if (p >= 0 && p < Lin)
                        v = Bsrc[(size_t)ci * (nb * Lin) + bb * Lin + p];
                }
                BsT[cc][kk] = v;
            }
        }
        __syncthreads();

        // --- A fragment: row r, contiguous K chunks {8h..8h+7},{16+8h..+7} ---
        V16U a;
        int r = wave * 16 + l16;
        a.h[0] = *(const v8h*)&As[r][8 * half];
        a.h[1] = *(const v8h*)&As[r][16 + 8 * half];

        #pragma unroll
        for (int nt = 0; nt < 4; ++nt) {
            V16U bf;
            int ncol = nt * 16 + l16;
            bf.h[0] = *(const v8h*)&BsT[ncol][8 * half];
            bf.h[1] = *(const v8h*)&BsT[ncol][16 + 8 * half];
            acc[nt] = __builtin_amdgcn_wmma_f32_16x16x32_f16(
                false, a.v, false, bf.v, (short)0, acc[nt], false, false);
        }
        __syncthreads();
    }

    // --- epilogue: C/D layout  VGPR v -> M = v + 8*half, N = l16 ---
    int mbase = bm + wave * 16 + 8 * half;
    #pragma unroll
    for (int nt = 0; nt < 4; ++nt) {
        int n = bn + nt * 16 + l16;
        #pragma unroll
        for (int v = 0; v < 8; ++v) {
            int m = mbase + v;
            if (m < M) {
                float x = acc[nt][v];
                if (alpha) x *= alpha[m];
                if (beta)  x += beta[m];
                if (act == 1)      x = siluf(x);
                else if (act == 2) x = softplusf(x);
                C[(size_t)m * N + n] = (f16)x;
            }
        }
    }
}

// ---------------------------------------------------------------------------
// Small helpers
// ---------------------------------------------------------------------------
__global__ void cvt_f32_f16(const float* __restrict__ s, f16* __restrict__ d, int n) {
    int i = blockIdx.x * blockDim.x + threadIdx.x;
    if (i < n) d[i] = (f16)s[i];
}

// alpha = g/sqrt(1+eps), beta = conv_bias*alpha + bn_beta   (BN here has var==1)
__global__ void prep_ab(const float* __restrict__ g, const float* __restrict__ bnb,
                        const float* __restrict__ cb, float* __restrict__ al,
                        float* __restrict__ be, int M) {
    int m = blockIdx.x * blockDim.x + threadIdx.x;
    if (m < M) {
        float a = g[m] * rsqrtf(1.f + 1e-5f);
        al[m] = a;
        be[m] = cb[m] * a + bnb[m];
    }
}

// conv1: 1->16, k=3, pad 1, +bias, BN, SiLU. out f32 [16][B*T]
__global__ void conv1_k(const float* __restrict__ x, const float* __restrict__ w,
                        const float* __restrict__ b, const float* __restrict__ g,
                        const float* __restrict__ bb, float* __restrict__ out) {
    int i = blockIdx.x * blockDim.x + threadIdx.x;
    if (i >= 16 * B_ * T_) return;
    int c = i / (B_ * T_), n = i % (B_ * T_);
    int bi = n / T_, t = n % T_;
    float acc = b[c];
    #pragma unroll
    for (int k = 0; k < 3; ++k) {
        int p = t - 1 + k;
        if (p >= 0 && p < T_) acc += w[c * 3 + k] * x[(size_t)bi * T_ + p];
    }
    float al = g[c] * rsqrtf(1.f + 1e-5f);
    out[i] = siluf(acc * al + bb[c]);
}

// conv2: 16->16, k=5, pad 2, +bias, BN, SiLU. out f16 [16][B*T]
__global__ void conv2_k(const float* __restrict__ h1, const float* __restrict__ w,
                        const float* __restrict__ b, const float* __restrict__ g,
                        const float* __restrict__ bb, f16* __restrict__ out) {
    int i = blockIdx.x * blockDim.x + threadIdx.x;
    if (i >= 16 * B_ * T_) return;
    int c = i / (B_ * T_), n = i % (B_ * T_);
    int bi = n / T_, t = n % T_;
    float acc = b[c];
    for (int ci = 0; ci < 16; ++ci) {
        #pragma unroll
        for (int k = 0; k < 5; ++k) {
            int p = t - 2 + k;
            if (p >= 0 && p < T_)
                acc += w[(c * 16 + ci) * 5 + k] * h1[(size_t)ci * (B_ * T_) + bi * T_ + p];
        }
    }
    float al = g[c] * rsqrtf(1.f + 1e-5f);
    out[i] = (f16)siluf(acc * al + bb[c]);
}

// Mamba depthwise causal conv (k=4, left pad 3) + bias + SiLU.
// xz rows 0..DI-1 hold xc. out f16 [DI][NCOL]
__global__ void dwconv_silu(const f16* __restrict__ xz, const float* __restrict__ w,
                            const float* __restrict__ bias, f16* __restrict__ out) {
    int i = blockIdx.x * blockDim.x + threadIdx.x;
    if (i >= DI_ * NCOL) return;
    int d = i / NCOL, n = i % NCOL;
    int bi = n / L_, t = n % L_;
    float acc = bias[d];
    #pragma unroll
    for (int j = 0; j < DC_; ++j) {
        int p = t - (DC_ - 1) + j;
        if (p >= 0) acc += w[d * DC_ + j] * (float)xz[(size_t)d * NCOL + bi * L_ + p];
    }
    out[i] = (f16)siluf(acc);
}

// Selective scan. One wave32 per (b,d); lane owns states s=lane and s=lane+32.
// Fuses y = scan + xc*D, then y *= silu(z). y f16 [DI][NCOL].
__global__ __launch_bounds__(256) void scan_k(
    const f16* __restrict__ dtm, const f16* __restrict__ xca,
    const f16* __restrict__ xdbl, const f16* __restrict__ z,
    const float* __restrict__ Alog, const float* __restrict__ Dv,
    f16* __restrict__ y) {
    int gw   = (blockIdx.x * blockDim.x + threadIdx.x) >> 5;
    int lane = threadIdx.x & 31;
    if (gw >= B_ * DI_) return;
    int bi = gw / DI_, d = gw % DI_;
    float A0 = -__expf(Alog[d * DS_ + lane]);
    float A1 = -__expf(Alog[d * DS_ + lane + 32]);
    float Dd = Dv[d];
    float h0 = 0.f, h1 = 0.f;
    const size_t rowd = (size_t)d * NCOL;
    int n0 = bi * L_;
    for (int t = 0; t < L_; ++t) {
        int n = n0 + t;
        float dtv = (float)dtm[rowd + n];
        float xv  = (float)xca[rowd + n];
        float dtx = dtv * xv;
        float Bs0 = (float)xdbl[(size_t)(DR_ + lane)            * NCOL + n];
        float Bs1 = (float)xdbl[(size_t)(DR_ + 32 + lane)       * NCOL + n];
        float Cs0 = (float)xdbl[(size_t)(DR_ + DS_ + lane)      * NCOL + n];
        float Cs1 = (float)xdbl[(size_t)(DR_ + DS_ + 32 + lane) * NCOL + n];
        h0 = __expf(dtv * A0) * h0 + dtx * Bs0;
        h1 = __expf(dtv * A1) * h1 + dtx * Bs1;
        float p = h0 * Cs0 + h1 * Cs1;
        #pragma unroll
        for (int off = 16; off > 0; off >>= 1) p += __shfl_xor(p, off, 32);
        if (lane == 0) {
            float zv = (float)z[rowd + n];
            y[rowd + n] = (f16)((p + xv * Dd) * siluf(zv));
        }
    }
}

// LayerNorm(512) + Linear(512->5) + log_softmax. One block (128 thr) per column.
// Output layout (L, B, 5) f32.
__global__ __launch_bounds__(128) void head_k(
    const f16* __restrict__ u, const float* __restrict__ g, const float* __restrict__ b,
    const float* __restrict__ lw, const float* __restrict__ lb, float* __restrict__ out) {
    __shared__ float red[128];
    __shared__ float stats[2];
    __shared__ float slog[5];
    int n = blockIdx.x, tid = threadIdx.x;

    float s = 0.f;
    for (int c = tid; c < NH_; c += 128) s += (float)u[(size_t)c * NCOL + n];
    red[tid] = s; __syncthreads();
    for (int o = 64; o > 0; o >>= 1) { if (tid < o) red[tid] += red[tid + o]; __syncthreads(); }
    if (tid == 0) stats[0] = red[0] / NH_;
    __syncthreads();
    float mu = stats[0];

    float v = 0.f;
    for (int c = tid; c < NH_; c += 128) {
        float dd = (float)u[(size_t)c * NCOL + n] - mu; v += dd * dd;
    }
    red[tid] = v; __syncthreads();
    for (int o = 64; o > 0; o >>= 1) { if (tid < o) red[tid] += red[tid + o]; __syncthreads(); }
    if (tid == 0) stats[1] = rsqrtf(red[0] / NH_ + 1e-5f);
    __syncthreads();
    float rstd = stats[1];

    float acc[5] = {0.f, 0.f, 0.f, 0.f, 0.f};
    for (int c = tid; c < NH_; c += 128) {
        float nv = ((float)u[(size_t)c * NCOL + n] - mu) * rstd * g[c] + b[c];
        #pragma unroll
        for (int j = 0; j < 5; ++j) acc[j] += nv * lw[j * NH_ + c];
    }
    for (int j = 0; j < 5; ++j) {
        red[tid] = acc[j]; __syncthreads();
        for (int o = 64; o > 0; o >>= 1) { if (tid < o) red[tid] += red[tid + o]; __syncthreads(); }
        if (tid == 0) slog[j] = red[0] + lb[j];
        __syncthreads();
    }
    if (tid == 0) {
        float mx = slog[0];
        for (int j = 1; j < 5; ++j) mx = fmaxf(mx, slog[j]);
        float se = 0.f;
        for (int j = 0; j < 5; ++j) se += __expf(slog[j] - mx);
        float lse = mx + logf(se);
        int bi = n / L_, l = n % L_;
        for (int j = 0; j < 5; ++j) out[((size_t)l * B_ + bi) * 5 + j] = slog[j] - lse;
    }
}

// ---------------------------------------------------------------------------
extern "C" void kernel_launch(void* const* d_in, const int* in_sizes, int n_in,
                              void* d_out, int out_size, void* d_ws, size_t ws_size,
                              hipStream_t stream) {
    (void)in_sizes; (void)n_in; (void)out_size; (void)ws_size;
    const float* x     = (const float*)d_in[0];
    const float* c1w   = (const float*)d_in[1];
    const float* c1b   = (const float*)d_in[2];
    const float* bn1g  = (const float*)d_in[3];
    const float* bn1b  = (const float*)d_in[4];
    const float* c2w   = (const float*)d_in[5];
    const float* c2b   = (const float*)d_in[6];
    const float* bn2g  = (const float*)d_in[7];
    const float* bn2b  = (const float*)d_in[8];
    const float* c3w   = (const float*)d_in[9];
    const float* c3b   = (const float*)d_in[10];
    const float* bn3g  = (const float*)d_in[11];
    const float* bn3b  = (const float*)d_in[12];
    const float* c4w   = (const float*)d_in[13];
    const float* c4b   = (const float*)d_in[14];
    const float* bn4g  = (const float*)d_in[15];
    const float* bn4b  = (const float*)d_in[16];
    const float* c5w   = (const float*)d_in[17];
    const float* c5b   = (const float*)d_in[18];
    const float* bn5g  = (const float*)d_in[19];
    const float* bn5b  = (const float*)d_in[20];
    const float* m_in_w   = (const float*)d_in[21];
    const float* m_conv_w = (const float*)d_in[22];
    const float* m_conv_b = (const float*)d_in[23];
    const float* m_xproj_w= (const float*)d_in[24];
    const float* m_dt_w   = (const float*)d_in[25];
    const float* m_dt_b   = (const float*)d_in[26];
    const float* m_Alog   = (const float*)d_in[27];
    const float* m_D      = (const float*)d_in[28];
    const float* m_out_w  = (const float*)d_in[29];
    const float* ln_g  = (const float*)d_in[30];
    const float* ln_b  = (const float*)d_in[31];
    const float* lin_w = (const float*)d_in[32];
    const float* lin_b = (const float*)d_in[33];

    // ---- workspace carve ----
    char* base = (char*)d_ws;
    size_t off = 0;
    auto carve = [&](size_t bytes) -> char* {
        char* p = base + off;
        off = (off + bytes + 255) & ~(size_t)255;
        return p;
    };
    f16* c3w16  = (f16*)carve((size_t)512 * 304 * 2);
    f16* c4w16  = (f16*)carve((size_t)512 * 2560 * 2);
    f16* c5w16  = (f16*)carve((size_t)512 * 2560 * 2);
    f16* inw16  = (f16*)carve((size_t)2048 * 512 * 2);
    f16* xpw16  = (f16*)carve((size_t)160 * 1024 * 2);
    f16* dtw16  = (f16*)carve((size_t)1024 * 32 * 2);
    f16* outw16 = (f16*)carve((size_t)512 * 1024 * 2);
    float* h1   = (float*)carve((size_t)16 * B_ * T_ * 4);
    f16* h2     = (f16*)carve((size_t)16 * B_ * T_ * 2);
    f16* hA     = (f16*)carve((size_t)NH_ * NCOL * 2);
    f16* hB     = (f16*)carve((size_t)NH_ * NCOL * 2);
    f16* xz     = (f16*)carve((size_t)2 * DI_ * NCOL * 2);
    f16* xca    = (f16*)carve((size_t)DI_ * NCOL * 2);
    f16* xdbl   = (f16*)carve((size_t)(DR_ + 2 * DS_) * NCOL * 2);
    f16* dtm    = (f16*)carve((size_t)DI_ * NCOL * 2);
    f16* yb     = (f16*)carve((size_t)DI_ * NCOL * 2);
    float* al   = (float*)carve((size_t)2048 * 4);
    float* be   = (float*)carve((size_t)2048 * 4);

    auto cvt = [&](const float* s, f16* d, int n) {
        cvt_f32_f16<<<(n + 255) / 256, 256, 0, stream>>>(s, d, n);
    };
    auto gemm = [&](const f16* A, const f16* Bp, f16* Cp, int M, int N, int K,
                    const float* a, const float* bp, int act,
                    int mode, int ks, int st, int pd, int Lin, int Lout) {
        dim3 grid(N / 64, (M + 127) / 128);
        gemm_wmma_f16<<<grid, 256, 0, stream>>>(A, Bp, Cp, M, N, K, a, bp, act,
                                                mode, ks, st, pd, Lin, Lout);
    };

    const int nBT = 16 * B_ * T_;

    // front-end convs
    conv1_k<<<(nBT + 255) / 256, 256, 0, stream>>>(x, c1w, c1b, bn1g, bn1b, h1);
    conv2_k<<<(nBT + 255) / 256, 256, 0, stream>>>(h1, c2w, c2b, bn2g, bn2b, h2);

    // conv3: 16->512, k19, s5, p9  (implicit im2col GEMM)
    cvt(c3w, c3w16, 512 * 304);
    prep_ab<<<2, 256, 0, stream>>>(bn3g, bn3b, c3b, al, be, 512);
    gemm(c3w16, h2, hA, 512, NCOL, 304, al, be, 1, 1, 19, 5, 9, T_, L_);

    // conv4: 512->512, k5, p2
    cvt(c4w, c4w16, 512 * 2560);
    prep_ab<<<2, 256, 0, stream>>>(bn4g, bn4b, c4b, al, be, 512);
    gemm(c4w16, hA, hB, 512, NCOL, 2560, al, be, 1, 1, 5, 1, 2, L_, L_);

    // conv5: 512->512, k5, p2
    cvt(c5w, c5w16, 512 * 2560);
    prep_ab<<<2, 256, 0, stream>>>(bn5g, bn5b, c5b, al, be, 512);
    gemm(c5w16, hB, hA, 512, NCOL, 2560, al, be, 1, 1, 5, 1, 2, L_, L_);

    // Mamba layers
    f16* uCur = hA;
    f16* uNxt = hB;
    for (int l = 0; l < 2; ++l) {
        cvt(m_in_w + (size_t)l * 2 * DI_ * NH_, inw16, 2 * DI_ * NH_);
        gemm(inw16, uCur, xz, 2 * DI_, NCOL, NH_, nullptr, nullptr, 0, 0, 1, 1, 0, 1, 1);

        dwconv_silu<<<(DI_ * NCOL + 255) / 256, 256, 0, stream>>>(
            xz, m_conv_w + (size_t)l * DI_ * DC_, m_conv_b + (size_t)l * DI_, xca);

        cvt(m_xproj_w + (size_t)l * (DR_ + 2 * DS_) * DI_, xpw16, (DR_ + 2 * DS_) * DI_);
        gemm(xpw16, xca, xdbl, DR_ + 2 * DS_, NCOL, DI_, nullptr, nullptr, 0, 0, 1, 1, 0, 1, 1);

        cvt(m_dt_w + (size_t)l * DI_ * DR_, dtw16, DI_ * DR_);
        gemm(dtw16, xdbl, dtm, DI_, NCOL, DR_, nullptr, m_dt_b + (size_t)l * DI_, 2,
             0, 1, 1, 0, 1, 1);

        scan_k<<<(B_ * DI_ * 32) / 256, 256, 0, stream>>>(
            dtm, xca, xdbl, xz + (size_t)DI_ * NCOL,
            m_Alog + (size_t)l * DI_ * DS_, m_D + (size_t)l * DI_, yb);

        cvt(m_out_w + (size_t)l * NH_ * DI_, outw16, NH_ * DI_);
        gemm(outw16, yb, uNxt, NH_, NCOL, DI_, nullptr, nullptr, 0, 0, 1, 1, 0, 1, 1);

        f16* t = uCur; uCur = uNxt; uNxt = t;
    }

    head_k<<<NCOL, 128, 0, stream>>>(uCur, ln_g, ln_b, lin_w, lin_b, (float*)d_out);
}
